// FasterMultiHeadAttention_18219251269687
// MI455X (gfx1250) — matrix operand order repaired and checked by reading, not compiled, not run on hardware
//
#include <hip/hip_runtime.h>

#define B_ 2
#define L_ 2048
#define E_ 1024
#define H_ 16
#define D_ 64

typedef __attribute__((ext_vector_type(16))) _Float16 v16h;
typedef __attribute__((ext_vector_type(8)))  _Float16 v8h;
typedef __attribute__((ext_vector_type(8)))  float    v8f;

static __device__ inline v16h make_frag(v8h lo, v8h hp) {
  return __builtin_shufflevector(lo, hp, 0,1,2,3,4,5,6,7,8,9,10,11,12,13,14,15);
}

// A-side fragment: 16x32 f16 tile (MxK). Rows contiguous along K in `src`,
// row stride `stride` halfs. lane<16: M=lane, K=k0+{0..7,16..23};
// lane>=16: M=lane-16, K=k0+{8..15,24..31}.  (ISA 7.12.2 16-bit A layout)
static __device__ inline v16h frag_a(const _Float16* src, int row0, int k0, int stride) {
  const int lane = threadIdx.x & 31;
  const int m  = lane & 15;
  const int hh = lane >> 4;
  const _Float16* p = src + (row0 + m) * stride + k0 + hh * 8;
  v8h lo = *(const v8h*)p;          // K = k0 + hh*8 + 0..7
  v8h hp = *(const v8h*)(p + 16);   // K = k0 + 16 + hh*8 + 0..7
  return make_frag(lo, hp);
}

// B-side fragment: 32x16 f16 tile (KxN). K rows contiguous along N in `src`,
// row stride `stride` halfs. lane holds column N=n0+(lane&15); K=k0+{0..7,16..23}
// for lane<16, K=k0+{8..15,24..31} for lane>=16 (same K order as frag_a).
static __device__ inline v16h frag_b(const _Float16* src, int n0, int k0, int stride) {
  const int lane = threadIdx.x & 31;
  const int n = n0 + (lane & 15);
  const int hh = lane >> 4;
  v16h r;
  #pragma unroll
  for (int s = 0; s < 8; ++s) {
    r[s]     = src[(k0 + 8 * hh + s) * stride + n];
    r[s + 8] = src[(k0 + 16 + 8 * hh + s) * stride + n];
  }
  return r;
}

static __device__ inline v8f wmma(v16h a, v16h b, v8f c) {
  return __builtin_amdgcn_wmma_f32_16x16x32_f16(false, a, false, b, (short)0, c, false, false);
}

// Async memory->LDS copy (16 bytes), tracked by ASYNCcnt. LDS destination is
// the low 32 bits of the generic shared-memory address; global address is a
// 64-bit VGPR pair.  (CDNA5 §10 / §15.18: GLOBAL_LOAD_ASYNC_TO_LDS_B128)
static __device__ inline void async_b128(void* lds, const void* g) {
  asm volatile("global_load_async_to_lds_b128 %0, %1, off"
               :: "v"((unsigned int)(unsigned long long)(uintptr_t)lds),
                  "v"((unsigned long long)(uintptr_t)g)
               : "memory");
}
static __device__ inline void wait_async0() {
  asm volatile("s_wait_asynccnt 0" ::: "memory");
}

// ---------------------------------------------------------------------------
// Kernel 1: fused Q/K/V projections. blockIdx.z selects which projection.
// y = x @ W^T (+bias, *0.125 for Q). Outputs f16 in WMMA-friendly layouts:
//   Q -> [b,h,l,d]   (A-side of QK^T)
//   K -> [b,h,d,l]   (B-side of QK^T: K-dim=d rows contiguous along keys)
//   V -> [b,h,l,d]   (B-side of PV:  K-dim=key rows contiguous along d)
// ---------------------------------------------------------------------------
__global__ __launch_bounds__(128) void proj_qkv(
    const float* __restrict__ q_in, const float* __restrict__ k_in,
    const float* __restrict__ v_in, const float* __restrict__ Wq,
    const float* __restrict__ Wk, const float* __restrict__ Wv,
    const float* __restrict__ bq,
    _Float16* __restrict__ Qh, _Float16* __restrict__ Kt,
    _Float16* __restrict__ Vh)
{
  __shared__ _Float16 As[64][40];  // [m][k], f16
  __shared__ _Float16 Ws[32][72];  // [k][n], f16 (transposed at stage-in)

  const int which = blockIdx.z;
  const float* X = (which == 0) ? q_in : ((which == 1) ? k_in : v_in);
  const float* W = (which == 0) ? Wq : ((which == 1) ? Wk : Wv);

  const int m0 = blockIdx.y * 64;
  const int n0 = blockIdx.x * 64;
  const int tid  = threadIdx.x;
  const int wave = tid >> 5;
  const int lane = tid & 31;
  const int wm = (wave & 1) * 32;
  const int wn = (wave >> 1) * 32;

  v8f zero = {0.f,0.f,0.f,0.f,0.f,0.f,0.f,0.f};
  v8f c[2][2];
  c[0][0] = zero; c[0][1] = zero; c[1][0] = zero; c[1][1] = zero;

  for (int k0 = 0; k0 < E_; k0 += 32) {
    __syncthreads();
    { // stage A: 64x32 f32 -> f16 LDS
      const int r  = tid >> 1;
      const int cb = (tid & 1) * 16;
      const float* g = X + (size_t)(m0 + r) * E_ + k0 + cb;
      alignas(16) _Float16 t[16];
      #pragma unroll
      for (int i = 0; i < 16; i += 4) {
        float4 f = *(const float4*)(g + i);
        t[i+0] = (_Float16)f.x; t[i+1] = (_Float16)f.y;
        t[i+2] = (_Float16)f.z; t[i+3] = (_Float16)f.w;
      }
      *(v8h*)&As[r][cb]     = *(const v8h*)&t[0];
      *(v8h*)&As[r][cb + 8] = *(const v8h*)&t[8];
      if (k0 + 32 < E_) __builtin_prefetch(g + 32, 0, 3);   // next A slab
    }
    { // stage W transposed: Ws[k][n] = W[n0+n][k0+k]
      const int n  = tid >> 1;
      const int kb = (tid & 1) * 16;
      const float* g = W + (size_t)(n0 + n) * E_ + k0 + kb;
      #pragma unroll
      for (int i = 0; i < 16; i += 4) {
        float4 f = *(const float4*)(g + i);
        Ws[kb+i+0][n] = (_Float16)f.x; Ws[kb+i+1][n] = (_Float16)f.y;
        Ws[kb+i+2][n] = (_Float16)f.z; Ws[kb+i+3][n] = (_Float16)f.w;
      }
      if (k0 + 32 < E_) __builtin_prefetch(g + 32, 0, 3);   // next W slab
    }
    __syncthreads();

    v16h a0 = frag_a(&As[0][0], wm,      0, 40);
    v16h a1 = frag_a(&As[0][0], wm + 16, 0, 40);
    v16h b0 = frag_b(&Ws[0][0], wn,      0, 72);
    v16h b1 = frag_b(&Ws[0][0], wn + 16, 0, 72);
    c[0][0] = wmma(a0, b0, c[0][0]);
    c[0][1] = wmma(a0, b1, c[0][1]);
    c[1][0] = wmma(a1, b0, c[1][0]);
    c[1][1] = wmma(a1, b1, c[1][1]);
  }

  // epilogue: C layout -> lane<16 rows r, lane>=16 rows r+8; col = lane&15
  const int hh = lane >> 4, ln = lane & 15;
  #pragma unroll
  for (int mi = 0; mi < 2; ++mi)
  #pragma unroll
  for (int nj = 0; nj < 2; ++nj)
  #pragma unroll
  for (int r = 0; r < 8; ++r) {
    const int gm = m0 + wm + mi * 16 + r + 8 * hh;   // global row  (b*L + l)
    const int gn = n0 + wn + nj * 16 + ln;           // global col  (h*D + d)
    float v = c[mi][nj][r];
    const int b = gm >> 11, l = gm & (L_ - 1);
    const int h = gn >> 6,  d = gn & (D_ - 1);
    if (which == 0) {
      v = (v + bq[gn]) * 0.125f;                     // fold 1/sqrt(D)
      Qh[(((size_t)(b * H_ + h)) * L_ + l) * D_ + d] = (_Float16)v;
    } else if (which == 1) {
      Kt[(((size_t)(b * H_ + h)) * D_ + d) * L_ + l] = (_Float16)v;
    } else {
      Vh[(((size_t)(b * H_ + h)) * L_ + l) * D_ + d] = (_Float16)v;
    }
  }
}

// ---------------------------------------------------------------------------
// Kernel 2: flash attention. Block = 64 query rows of one (b,h); 4 waves,
// each owns a 16-row tile. K/V tiles staged via async memory->LDS copies.
// ---------------------------------------------------------------------------
__global__ __launch_bounds__(128) void attn(
    const _Float16* __restrict__ Qh, const _Float16* __restrict__ Kt,
    const _Float16* __restrict__ Vh, _Float16* __restrict__ Ah)
{
  __shared__ _Float16 Kts[64][40];    // [d][key] for the current 32-key block
  __shared__ _Float16 Vs[32][72];     // [key][d]
  __shared__ _Float16 Pb[4][16][40];  // per-wave P transpose staging

  const int tid  = threadIdx.x;
  const int wave = tid >> 5;
  const int lane = tid & 31;
  const int hh = lane >> 4, ln = lane & 15;
  const int bh = blockIdx.y;                   // b*H + h
  const int q0 = blockIdx.x * 64 + wave * 16;  // this wave's query tile

  const _Float16* Qp = Qh + (size_t)bh * L_ * D_;
  const _Float16* Kp = Kt + (size_t)bh * D_ * L_;
  const _Float16* Vp = Vh + (size_t)bh * L_ * D_;

  // Q fragments resident in registers for the whole loop (d = 0..31, 32..63)
  const v16h aq0 = frag_a(Qp + (size_t)q0 * D_, 0, 0,  D_);
  const v16h aq1 = frag_a(Qp + (size_t)q0 * D_, 0, 32, D_);

  v8f zero = {0.f,0.f,0.f,0.f,0.f,0.f,0.f,0.f};
  v8f o[4]; float m[8], lsum[8];
  #pragma unroll
  for (int t = 0; t < 4; ++t) o[t] = zero;
  #pragma unroll
  for (int r = 0; r < 8; ++r) { m[r] = -1e30f; lsum[r] = 0.f; }

  for (int j = 0; j < L_ / 32; ++j) {
    const int key0 = j * 32;
    __syncthreads();
    { // async stage K^T block: 64 d-rows x 32 keys (ASYNCcnt path)
      const int d  = tid >> 1;
      const int cb = (tid & 1) * 16;
      const _Float16* g = Kp + (size_t)d * L_ + key0 + cb;
      async_b128(&Kts[d][cb],     g);
      async_b128(&Kts[d][cb + 8], g + 8);
    }
    { // async stage V block: 32 key-rows x 64 d
      const int r  = tid >> 2;
      const int cb = (tid & 3) * 16;
      const _Float16* g = Vp + (size_t)(key0 + r) * D_ + cb;
      async_b128(&Vs[r][cb],     g);
      async_b128(&Vs[r][cb + 8], g + 8);
    }
    wait_async0();
    __syncthreads();

    // scores: 16 q x 32 keys, contraction over d=64 (two chained WMMAs each)
    v8f s0 = zero, s1 = zero;
    s0 = wmma(aq0, frag_b(&Kts[0][0], 0,  0,  40), s0);
    s0 = wmma(aq1, frag_b(&Kts[0][0], 0,  32, 40), s0);
    s1 = wmma(aq0, frag_b(&Kts[0][0], 16, 0,  40), s1);
    s1 = wmma(aq1, frag_b(&Kts[0][0], 16, 32, 40), s1);

    // online softmax per row (row = r + 8*hh; 16 lanes of a half share a row)
    #pragma unroll
    for (int r = 0; r < 8; ++r) {
      float rmax = fmaxf(s0[r], s1[r]);
      rmax = fmaxf(rmax, __shfl_xor(rmax, 1, 32));
      rmax = fmaxf(rmax, __shfl_xor(rmax, 2, 32));
      rmax = fmaxf(rmax, __shfl_xor(rmax, 4, 32));
      rmax = fmaxf(rmax, __shfl_xor(rmax, 8, 32));
      const float mn   = fmaxf(m[r], rmax);
      const float corr = __expf(m[r] - mn);
      m[r] = mn;
      const float p0 = __expf(s0[r] - mn);
      const float p1 = __expf(s1[r] - mn);
      float rs = p0 + p1;
      rs += __shfl_xor(rs, 1, 32);
      rs += __shfl_xor(rs, 2, 32);
      rs += __shfl_xor(rs, 4, 32);
      rs += __shfl_xor(rs, 8, 32);
      lsum[r] = lsum[r] * corr + rs;
      #pragma unroll
      for (int t = 0; t < 4; ++t) o[t][r] *= corr;
      // C-layout -> LDS (A-layout source for the PV matmul)
      Pb[wave][r + 8 * hh][ln]      = (_Float16)p0;
      Pb[wave][r + 8 * hh][16 + ln] = (_Float16)p1;
    }
    // wave-internal LDS RAW: same-wave DS ops are in-order; wait them out
    asm volatile("s_wait_dscnt 0" ::: "memory");

    const v16h pa = frag_a(&Pb[wave][0][0], 0, 0, 40);
    #pragma unroll
    for (int t = 0; t < 4; ++t)
      o[t] = wmma(pa, frag_b(&Vs[0][0], t * 16, 0, 72), o[t]);
  }

  // normalize and store attention output as [b*L + q][h*D + d] f16
  const int b = bh >> 4, h = bh & (H_ - 1);
  #pragma unroll
  for (int r = 0; r < 8; ++r) {
    const float inv = 1.0f / lsum[r];
    const size_t row = (size_t)b * L_ + q0 + r + 8 * hh;
    #pragma unroll
    for (int t = 0; t < 4; ++t) {
      const int col = h * D_ + t * 16 + ln;
      Ah[row * E_ + col] = (_Float16)(o[t][r] * inv);
    }
  }
}

// ---------------------------------------------------------------------------
// Kernel 3: output projection, f16 A (attn out) x f16 Wo -> fp32 d_out (+bo)
// A tile staged with async memory->LDS copies (already f16, no conversion).
// ---------------------------------------------------------------------------
__global__ __launch_bounds__(128) void proj_out(
    const _Float16* __restrict__ Ain, const float* __restrict__ Wo,
    const float* __restrict__ bo, float* __restrict__ Out)
{
  __shared__ _Float16 As[64][40];
  __shared__ _Float16 Ws[32][72];

  const int m0 = blockIdx.y * 64;
  const int n0 = blockIdx.x * 64;
  const int tid  = threadIdx.x;
  const int wave = tid >> 5;
  const int lane = tid & 31;
  const int wm = (wave & 1) * 32;
  const int wn = (wave >> 1) * 32;

  v8f zero = {0.f,0.f,0.f,0.f,0.f,0.f,0.f,0.f};
  v8f c[2][2];
  c[0][0] = zero; c[0][1] = zero; c[1][0] = zero; c[1][1] = zero;

  for (int k0 = 0; k0 < E_; k0 += 32) {
    __syncthreads();
    { // async stage A (already f16)
      const int r  = tid >> 1;
      const int cb = (tid & 1) * 16;
      const _Float16* g = Ain + (size_t)(m0 + r) * E_ + k0 + cb;
      async_b128(&As[r][cb],     g);
      async_b128(&As[r][cb + 8], g + 8);
      if (k0 + 32 < E_) __builtin_prefetch(g + 32, 0, 3);
    }
    { // stage Wo transposed, f32 -> f16
      const int n  = tid >> 1;
      const int kb = (tid & 1) * 16;
      const float* g = Wo + (size_t)(n0 + n) * E_ + k0 + kb;
      #pragma unroll
      for (int i = 0; i < 16; i += 4) {
        float4 f = *(const float4*)(g + i);
        Ws[kb+i+0][n] = (_Float16)f.x; Ws[kb+i+1][n] = (_Float16)f.y;
        Ws[kb+i+2][n] = (_Float16)f.z; Ws[kb+i+3][n] = (_Float16)f.w;
      }
      if (k0 + 32 < E_) __builtin_prefetch(g + 32, 0, 3);
    }
    wait_async0();
    __syncthreads();

    v16h a0 = frag_a(&As[0][0], wm,      0, 40);
    v16h a1 = frag_a(&As[0][0], wm + 16, 0, 40);
    v16h b0 = frag_b(&Ws[0][0], wn,      0, 72);
    v16h b1 = frag_b(&Ws[0][0], wn + 16, 0, 72);
    c[0][0] = wmma(a0, b0, c[0][0]);
    c[0][1] = wmma(a0, b1, c[0][1]);
    c[1][0] = wmma(a1, b0, c[1][0]);
    c[1][1] = wmma(a1, b1, c[1][1]);
  }

  const int hh = lane >> 4, ln = lane & 15;
  #pragma unroll
  for (int mi = 0; mi < 2; ++mi)
  #pragma unroll
  for (int nj = 0; nj < 2; ++nj)
  #pragma unroll
  for (int r = 0; r < 8; ++r) {
    const int gm = m0 + wm + mi * 16 + r + 8 * hh;
    const int gn = n0 + wn + nj * 16 + ln;
    Out[(size_t)gm * E_ + gn] = c[mi][nj][r] + bo[gn];
  }
}

extern "C" void kernel_launch(void* const* d_in, const int* in_sizes, int n_in,
                              void* d_out, int out_size, void* d_ws, size_t ws_size,
                              hipStream_t stream) {
  const float* q  = (const float*)d_in[0];
  const float* k  = (const float*)d_in[1];
  const float* v  = (const float*)d_in[2];
  const float* Wq = (const float*)d_in[3];
  const float* bq = (const float*)d_in[4];
  const float* Wk = (const float*)d_in[5];
  const float* Wv = (const float*)d_in[6];
  const float* Wo = (const float*)d_in[7];
  const float* bo = (const float*)d_in[8];
  float* out = (float*)d_out;

  const size_t perTensor = (size_t)B_ * H_ * L_ * D_;  // 4M halfs = 8 MB
  _Float16* Qh = (_Float16*)d_ws;
  _Float16* Kt = Qh + perTensor;
  _Float16* Vh = Kt + perTensor;
  _Float16* Ah = Vh + perTensor;

  dim3 blk(128);
  proj_qkv<<<dim3(E_ / 64, (B_ * L_) / 64, 3), blk, 0, stream>>>(
      q, k, v, Wq, Wk, Wv, bq, Qh, Kt, Vh);
  attn<<<dim3(L_ / 64, B_ * H_), blk, 0, stream>>>(Qh, Kt, Vh, Ah);
  proj_out<<<dim3(E_ / 64, (B_ * L_) / 64), blk, 0, stream>>>(Ah, Wo, bo, out);
}
